// GCN_8074538516509
// MI455X (gfx1250) — matrix operand order repaired and checked by reading, not compiled, not run on hardware
//
#include <hip/hip_runtime.h>
#include <hip/hip_bf16.h>
#include <math.h>

// ---------------- types for WMMA ----------------
typedef __attribute__((ext_vector_type(16))) _Float16 v16h;
typedef __attribute__((ext_vector_type(8)))  _Float16 v8h;
typedef __attribute__((ext_vector_type(8)))  float    v8f;

#define F_IN 128
#define HIDW 64   // hidden width == padded output width of every GEMM
#define NCLS 40

// ---------------- small elementwise kernels ----------------
__global__ void k_fill_f32(float* p, float v, int n) {
    int i = blockIdx.x * blockDim.x + threadIdx.x;
    if (i < n) p[i] = v;
}

__global__ void k_zero_f32(float* p, long n) {
    long i = (long)blockIdx.x * blockDim.x + threadIdx.x;
    if (i < n) p[i] = 0.0f;
}

__global__ void k_deg(const int* __restrict__ dst, float* __restrict__ degf, int E) {
    int e = blockIdx.x * blockDim.x + threadIdx.x;
    if (e < E) atomicAdd(&degf[dst[e]], 1.0f);
}

__global__ void k_rsqrt(const float* __restrict__ degf, float* __restrict__ dis, int n) {
    int i = blockIdx.x * blockDim.x + threadIdx.x;
    if (i < n) dis[i] = rsqrtf(degf[i]);   // deg >= 1 always (self loops)
}

__global__ void k_f32_to_f16(const float* __restrict__ in, _Float16* __restrict__ out, long n) {
    long i = (long)blockIdx.x * blockDim.x + threadIdx.x;
    if (i < n) out[i] = (_Float16)in[i];
}

// pad W3 (64 x 40) to (64 x 64) f16, zero-filled columns 40..63
__global__ void k_w3_pad(const float* __restrict__ W3, _Float16* __restrict__ out) {
    int i = blockIdx.x * blockDim.x + threadIdx.x;   // 64*64 threads
    if (i < 64 * 64) {
        int r = i >> 6, c = i & 63;
        out[i] = (c < NCLS) ? (_Float16)W3[r * NCLS + c] : (_Float16)0.0f;
    }
}

// ---------------- WMMA GEMM:  D[M x 64] = A[M x K] * B[K x 64] ----------------
// A row-major f16, B row-major f16 (staged to LDS in fragment order), D f32.
// One wave computes a 16-row x 64-col output strip with 4 accumulators.
template<int K>
__global__ __launch_bounds__(256) void k_gemm_wmma(const _Float16* __restrict__ A,
                                                   const _Float16* __restrict__ Bm,
                                                   float* __restrict__ D, int M) {
    constexpr int KT = K / 32;                 // k-steps of 32
    __shared__ _Float16 lds[K * 64];           // B in per-lane fragment order

    const int t = threadIdx.x;
    // Cooperative fill of B fragments.
    // Fragment (kt, nt), lane l, half i holds B[kt*32 + (l>=16?8:0) + (i<8?i:i+8)][nt*16 + (l&15)]
    // stored contiguously: lds[((kt*4+nt)*32 + l)*16 + i]  -> 32B per lane -> 2x ds_load_b128
    for (int idx = t; idx < K * 64; idx += 256) {
        int i    = idx & 15;
        int lane = (idx >> 4) & 31;
        int frag = idx >> 9;
        int kt   = frag >> 2, nt = frag & 3;
        int kk   = kt * 32 + ((lane & 16) ? 8 : 0) + (i < 8 ? i : i + 8);
        int col  = nt * 16 + (lane & 15);
        lds[idx] = Bm[kk * 64 + col];
    }
    __syncthreads();

    const int lane   = t & 31;
    const int waveId = t >> 5;
    const int tile   = blockIdx.x * 8 + waveId;
    const int tiles  = M >> 4;                 // M is a multiple of 16 (100000 = 6250*16)
    if (tile >= tiles) return;

    // A fragment addressing (ISA 7.12.2, 16-bit A 16x32):
    // lanes 0-15: row = lane,    halves = K {0..7, 16..23}
    // lanes 16-31: row = lane-16, halves = K {8..15, 24..31}
    const int arow = tile * 16 + (lane & 15);
    const int koff = (lane & 16) ? 8 : 0;
    const _Float16* aBase = A + (size_t)arow * K + koff;

    v8f z = {0, 0, 0, 0, 0, 0, 0, 0};
    v8f acc[4] = {z, z, z, z};

#pragma unroll
    for (int kt = 0; kt < KT; ++kt) {
        v8h alo = *reinterpret_cast<const v8h*>(aBase + kt * 32);        // K +0..7 (or +8..15)
        v8h ahi = *reinterpret_cast<const v8h*>(aBase + kt * 32 + 16);   // K +16..23 (or +24..31)
        v16h a = __builtin_shufflevector(alo, ahi, 0, 1, 2, 3, 4, 5, 6, 7,
                                                   8, 9, 10, 11, 12, 13, 14, 15);
#pragma unroll
        for (int nt = 0; nt < 4; ++nt) {
            const v16h b = *reinterpret_cast<const v16h*>(&lds[((kt * 4 + nt) * 32 + lane) * 16]);
            acc[nt] = __builtin_amdgcn_wmma_f32_16x16x32_f16(
                false, a, false, b, (short)0, acc[nt], false, false);
        }
    }

    // C/D layout: VGPR r -> row r for lanes 0-15, row r+8 for lanes 16-31; col = lane&15
    const int rbase = tile * 16 + ((lane & 16) ? 8 : 0);
    const int col   = lane & 15;
#pragma unroll
    for (int nt = 0; nt < 4; ++nt) {
#pragma unroll
        for (int r = 0; r < 8; ++r) {
            D[(size_t)(rbase + r) * 64 + nt * 16 + col] = acc[nt][r];
        }
    }
}

// ---------------- edge scatter: Agg[dst] += T[src] * norm ----------------
// one thread per (edge, 4-feature chunk); 16 chunks cover 64 features.
// edges [0,E) come from edge_index; edges [E, E+N) are self loops.
__global__ void k_scatter(const int* __restrict__ src, const int* __restrict__ dst,
                          const float* __restrict__ dis, const float* __restrict__ T,
                          float* __restrict__ Agg, int E, int Nn) {
    long tid = (long)blockIdx.x * blockDim.x + threadIdx.x;
    long e = tid >> 4;
    int  c = (int)(tid & 15) * 4;
    if (e >= (long)E + Nn) return;
    int s, d; float w;
    if (e < E) {
        s = src[e]; d = dst[e];
        w = dis[s] * dis[d];
    } else {
        s = d = (int)(e - E);
        float r = dis[s];
        w = r * r;                       // 1/deg
    }
    const float4 tv = *reinterpret_cast<const float4*>(T + (size_t)s * 64 + c);
    float* o = Agg + (size_t)d * 64 + c;
    atomicAdd(o + 0, tv.x * w);
    atomicAdd(o + 1, tv.y * w);
    atomicAdd(o + 2, tv.z * w);
    atomicAdd(o + 3, tv.w * w);
}

// ---------------- bias + swish + convert to f16 ----------------
__global__ void k_act(const float* __restrict__ Agg, const float* __restrict__ bias,
                      _Float16* __restrict__ H, long total) {
    long i = (long)blockIdx.x * blockDim.x + threadIdx.x;
    if (i >= total) return;
    float v = Agg[i] + bias[i & 63];
    float s = v / (1.0f + __expf(-v));   // swish = v * sigmoid(v)
    H[i] = (_Float16)s;
}

// ---------------- final: logits = Agg + b3 (cols 0..39), log_softmax ----------------
// one wave32 per row: lane handles col=lane and (lane<8) col=lane+32
__global__ void k_logsoftmax(const float* __restrict__ Agg, const float* __restrict__ b3,
                             float* __restrict__ out, int Nn) {
    int row  = blockIdx.x * (blockDim.x >> 5) + (threadIdx.x >> 5);
    int lane = threadIdx.x & 31;
    if (row >= Nn) return;
    const float* a = Agg + (size_t)row * 64;
    float v0 = a[lane] + b3[lane];
    float v1 = (lane < 8) ? (a[32 + lane] + b3[32 + lane]) : -3.0e38f;
    float m = fmaxf(v0, v1);
#pragma unroll
    for (int o = 16; o > 0; o >>= 1) m = fmaxf(m, __shfl_xor(m, o, 32));
    float e = __expf(v0 - m) + ((lane < 8) ? __expf(v1 - m) : 0.0f);
#pragma unroll
    for (int o = 16; o > 0; o >>= 1) e += __shfl_xor(e, o, 32);
    float lse = m + __logf(e);
    float* orow = out + (size_t)row * NCLS;
    orow[lane] = v0 - lse;
    if (lane < 8) orow[32 + lane] = v1 - lse;
}

// ---------------- host launch ----------------
extern "C" void kernel_launch(void* const* d_in, const int* in_sizes, int n_in,
                              void* d_out, int out_size, void* d_ws, size_t ws_size,
                              hipStream_t stream) {
    (void)n_in; (void)out_size; (void)ws_size;

    const float* x  = (const float*)d_in[0];
    const int*   ei = (const int*)d_in[1];
    const float* W1 = (const float*)d_in[2];
    const float* b1 = (const float*)d_in[3];
    const float* W2 = (const float*)d_in[4];
    const float* b2 = (const float*)d_in[5];
    const float* W3 = (const float*)d_in[6];
    const float* b3 = (const float*)d_in[7];
    float*       out = (float*)d_out;

    const int Nn = in_sizes[0] / F_IN;     // 100000
    const int E  = in_sizes[1] / 2;        // 1600000
    const int* src = ei;                   // edge_index row 0
    const int* dst = ei + E;               // edge_index row 1

    // workspace carve-out (256B aligned)
    char* ws = (char*)d_ws;
    size_t off = 0;
    auto carve = [&](size_t bytes) -> void* {
        off = (off + 255) & ~(size_t)255;
        void* p = ws + off;
        off += bytes;
        return p;
    };
    float*     degf = (float*)carve((size_t)Nn * 4);
    float*     dis  = (float*)carve((size_t)Nn * 4);
    _Float16*  Xf   = (_Float16*)carve((size_t)Nn * F_IN * 2);
    _Float16*  W1f  = (_Float16*)carve((size_t)F_IN * 64 * 2);
    _Float16*  W2f  = (_Float16*)carve((size_t)64 * 64 * 2);
    _Float16*  W3f  = (_Float16*)carve((size_t)64 * 64 * 2);
    float*     T    = (float*)carve((size_t)Nn * 64 * 4);
    float*     Agg  = (float*)carve((size_t)Nn * 64 * 4);
    _Float16*  Hf   = (_Float16*)carve((size_t)Nn * 64 * 2);

    const int TPB = 256;
    auto blk = [](long n, int tpb) { return (int)((n + tpb - 1) / tpb); };

    // ---- degree / norm ----
    k_fill_f32<<<blk(Nn, TPB), TPB, 0, stream>>>(degf, 1.0f, Nn);           // self loop
    k_deg<<<blk(E, TPB), TPB, 0, stream>>>(dst, degf, E);
    k_rsqrt<<<blk(Nn, TPB), TPB, 0, stream>>>(degf, dis, Nn);

    // ---- precision staging ----
    k_f32_to_f16<<<blk((long)Nn * F_IN, TPB), TPB, 0, stream>>>(x, Xf, (long)Nn * F_IN);
    k_f32_to_f16<<<blk(F_IN * 64, TPB), TPB, 0, stream>>>(W1, W1f, F_IN * 64);
    k_f32_to_f16<<<blk(64 * 64, TPB), TPB, 0, stream>>>(W2, W2f, 64 * 64);
    k_w3_pad<<<blk(64 * 64, TPB), TPB, 0, stream>>>(W3, W3f);

    const int tiles      = Nn / 16;                    // 6250
    const int gemmBlocks = (tiles + 7) / 8;            // 8 waves per block
    const long aggElems  = (long)Nn * 64;
    const long scatThr   = ((long)E + Nn) * 16;

    // ---- layer 1 ----
    k_gemm_wmma<F_IN><<<gemmBlocks, TPB, 0, stream>>>(Xf, W1f, T, Nn);
    k_zero_f32<<<blk(aggElems, TPB), TPB, 0, stream>>>(Agg, aggElems);
    k_scatter<<<blk(scatThr, TPB), TPB, 0, stream>>>(src, dst, dis, T, Agg, E, Nn);
    k_act<<<blk(aggElems, TPB), TPB, 0, stream>>>(Agg, b1, Hf, aggElems);

    // ---- layer 2 ----
    k_gemm_wmma<HIDW><<<gemmBlocks, TPB, 0, stream>>>(Hf, W2f, T, Nn);
    k_zero_f32<<<blk(aggElems, TPB), TPB, 0, stream>>>(Agg, aggElems);
    k_scatter<<<blk(scatThr, TPB), TPB, 0, stream>>>(src, dst, dis, T, Agg, E, Nn);
    k_act<<<blk(aggElems, TPB), TPB, 0, stream>>>(Agg, b2, Hf, aggElems);

    // ---- layer 3 ----
    k_gemm_wmma<HIDW><<<gemmBlocks, TPB, 0, stream>>>(Hf, W3f, T, Nn);
    k_zero_f32<<<blk(aggElems, TPB), TPB, 0, stream>>>(Agg, aggElems);
    k_scatter<<<blk(scatThr, TPB), TPB, 0, stream>>>(src, dst, dis, T, Agg, E, Nn);

    // ---- log softmax over 40 classes ----
    k_logsoftmax<<<blk((long)Nn * 32, TPB), TPB, 0, stream>>>(Agg, b3, out, Nn);
}
